// tcn_unit_attention_38208029065262
// MI455X (gfx1250) — matrix operand order repaired and checked by reading, not compiled, not run on hardware
//
#include <hip/hip_runtime.h>
#include <hip/hip_bf16.h>
#include <cstdint>

// ---------------------------------------------------------------------------
// TCN attention unit for MI455X (gfx1250): bf16 WMMA, f32 accumulate.
// GEMMs use 64x64 blocks per 4-wave workgroup with DOUBLE-BUFFERED
// GLOBAL_LOAD_ASYNC_TO_LDS_B128 staging (ASYNCcnt + split barriers).
// Attention processes 32 query rows per wave with rel_to_abs fused as a
// scatter-add into the logits LDS tile.
// ---------------------------------------------------------------------------

typedef __attribute__((ext_vector_type(16))) __bf16 v16bf;
typedef __attribute__((ext_vector_type(8)))  float  v8f;

#define DEV __device__ __forceinline__

constexpr int Nn  = 8;
constexpr int Cc  = 256;
constexpr int Tt  = 256;
constexpr int Vv  = 25;
constexpr int NHh = 8;
constexpr int DKH = 32;
constexpr int Bb  = Nn * Vv;     // 200
constexpr int O3  = 768;         // 2*DK + DV

DEV unsigned short f2bf(float f) {
  unsigned int u = __builtin_bit_cast(unsigned int, f);
  u += 0x7fffu + ((u >> 16) & 1u);          // round-to-nearest-even
  return (unsigned short)(u >> 16);
}

union FragU { uint4 q[2]; v16bf v; };
union AccU  { v8f v; float f[8]; };

// A fragment (16x32, bf16), source row-major [M][K], K contiguous.
// Lane l: row = l&15; halves 0..7 = K {8*(l>>4) .. +7}; halves 8..15 at +16.
DEV v16bf load_a_frag(const unsigned short* tile, int ld, int lane) {
  const unsigned short* p = tile + (lane & 15) * ld + ((lane >> 4) << 3);
  FragU u;
  u.q[0] = *(const uint4*)(p);
  u.q[1] = *(const uint4*)(p + 16);
  return u.v;
}

// B fragment (32x16, bf16), source is B^T row-major [N][K], K contiguous.
// Lane l: col = l&15; halves 0..7 = K {8*(l>>4) .. +7}; halves 8..15 at +16.
DEV v16bf load_b_frag(const unsigned short* tile, int ld, int lane) {
  const unsigned short* p = tile + (lane & 15) * ld + ((lane >> 4) << 3);
  FragU u;
  u.q[0] = *(const uint4*)(p);
  u.q[1] = *(const uint4*)(p + 16);
  return u.v;
}

DEV v8f wmma_bf(v16bf a, v16bf b, v8f c) {
  return __builtin_amdgcn_wmma_f32_16x16x32_bf16(
      false, a, false, b, (short)0, c, false, false);
}

// ---- CDNA5 async global->LDS (no VGPR staging, tracked by ASYNCcnt) -------
// Generic pointers to LDS are aperture|offset; low 32 bits are the LDS byte
// address the async instruction's VDST operand expects.
DEV unsigned lds_off(const void* p) {
  return (unsigned)(unsigned long long)p;
}
DEV void async_ld_b128(unsigned lds_byte_addr, const void* gsrc) {
  asm volatile("global_load_async_to_lds_b128 %0, %1, off"
               :: "v"(lds_byte_addr), "v"((unsigned long long)gsrc)
               : "memory");
}
DEV void wait_async_le4() { asm volatile("s_wait_asynccnt 4" ::: "memory"); }
DEV void wait_async_0()   { asm volatile("s_wait_asynccnt 0" ::: "memory"); }

// ---------------------------------------------------------------------------
// K0a: pack x (N,C,T,V) f32  ->  xT[b][t][c] bf16   (b = n*V + v)
// ---------------------------------------------------------------------------
__global__ void k_pack_x(const float* __restrict__ x, unsigned short* __restrict__ xT) {
  int idx = blockIdx.x * 256 + threadIdx.x;        // total = Bb*Tt*Cc
  int c = idx & (Cc - 1);
  int t = (idx >> 8) & (Tt - 1);
  int b = idx >> 16;
  int n = b / Vv, v = b - n * Vv;
  size_t xi = (((size_t)n * Cc + c) * Tt + t) * Vv + v;
  xT[idx] = f2bf(x[xi]);
}

// ---------------------------------------------------------------------------
// K0b: pack weights: qkv_w (768x256), attn_w (256x256), key_rel (511x32 -> 512x32)
// ---------------------------------------------------------------------------
__global__ void k_pack_w(const float* __restrict__ qkv_w,
                         const float* __restrict__ attn_w,
                         const float* __restrict__ key_rel,
                         unsigned short* __restrict__ qkvw_bf,
                         unsigned short* __restrict__ attnw_bf,
                         unsigned short* __restrict__ relw_bf) {
  int idx = blockIdx.x * 256 + threadIdx.x;
  if (idx < O3 * Cc) {
    qkvw_bf[idx] = f2bf(qkv_w[idx]);
  } else if (idx < O3 * Cc + Cc * Cc) {
    int i = idx - O3 * Cc;
    attnw_bf[i] = f2bf(attn_w[i]);
  } else if (idx < O3 * Cc + Cc * Cc + 512 * DKH) {
    int i = idx - O3 * Cc - Cc * Cc;
    int m = i >> 5;
    relw_bf[i] = (m < 2 * Tt - 1) ? f2bf(key_rel[i]) : (unsigned short)0;
  }
}

// ---------------------------------------------------------------------------
// Staged 64x64x256 GEMM body: 4 waves. Per 32-wide K step each thread owns a
// 32B segment of As and Bs; segments are filled with async-to-LDS transfers,
// double-buffered so the next step's staging overlaps this step's WMMAs.
// Wave w computes rows [16w,16w+16) x all 64 cols (4 accumulators).
// ---------------------------------------------------------------------------
DEV void gemm64x64(const unsigned short* __restrict__ A,   // [64][256] rows o
                   const unsigned short* __restrict__ Bt,  // [64][256] rows t
                   unsigned short* As, unsigned short* Bs, // LDS 2*64*32 each
                   int tid, int lane, int w, v8f acc[4]) {
  const int srow = tid >> 1, shalf = (tid & 1) << 4;   // 32B segment per thread
  const unsigned short* ga = A  + srow * Cc + shalf;
  const unsigned short* gb = Bt + srow * Cc + shalf;
  const unsigned la0 = lds_off(As + srow * 32 + shalf);
  const unsigned lb0 = lds_off(Bs + srow * 32 + shalf);

  // prologue: stage K-step 0 into buffer 0
  async_ld_b128(la0,      ga);
  async_ld_b128(la0 + 16, ga + 8);
  async_ld_b128(lb0,      gb);
  async_ld_b128(lb0 + 16, gb + 8);

#pragma unroll
  for (int kk = 0; kk < 8; ++kk) {
    const int cur = kk & 1;
    if (kk < 7) {                                      // stage next into other buf
      const unsigned buf = (cur ^ 1) * 4096;           // 64*32 bf16 = 4KB
      const unsigned short* gan = ga + (kk + 1) * 32;
      const unsigned short* gbn = gb + (kk + 1) * 32;
      async_ld_b128(la0 + buf,      gan);
      async_ld_b128(la0 + buf + 16, gan + 8);
      async_ld_b128(lb0 + buf,      gbn);
      async_ld_b128(lb0 + buf + 16, gbn + 8);
      wait_async_le4();                                // current buffer landed
    } else {
      wait_async_0();
    }
    __syncthreads();                                   // publish LDS to all waves
    const unsigned short* Asc = As + cur * 2048;
    const unsigned short* Bsc = Bs + cur * 2048;
    v16bf a = load_a_frag(Asc + w * 16 * 32, 32, lane);
#pragma unroll
    for (int nt = 0; nt < 4; ++nt) {
      v16bf bb = load_b_frag(Bsc + nt * 16 * 32, 32, lane);
      acc[nt] = wmma_bf(a, bb, acc[nt]);
    }
    __syncthreads();                                   // reads done before rewrite
  }
}

// ---------------------------------------------------------------------------
// K1: QKV projection.  qkv[b,o,t] = sum_c qkv_w[o,c]*xT[b,t,c] + qkv_b[o]
// grid (Bb, 12, 4), 128 threads. Scatter q/k: [b][h][t][d], v: [b][h][d][t].
// ---------------------------------------------------------------------------
__global__ void k_qkv_gemm(const unsigned short* __restrict__ qkvw_bf,
                           const unsigned short* __restrict__ xT,
                           const float* __restrict__ qkv_b,
                           unsigned short* __restrict__ qT,
                           unsigned short* __restrict__ kT,
                           unsigned short* __restrict__ vD) {
  __shared__ unsigned short As[2 * 64 * 32];
  __shared__ unsigned short Bs[2 * 64 * 32];
  const int tid = threadIdx.x, lane = tid & 31, w = tid >> 5;
  const int b = blockIdx.x, o0 = blockIdx.y * 64, t0 = blockIdx.z * 64;

  v8f acc[4] = {};
  gemm64x64(qkvw_bf + (size_t)o0 * Cc, xT + ((size_t)b * Tt + t0) * Cc,
            As, Bs, tid, lane, w, acc);

  const int tl = lane & 15, ob = 8 * (lane >> 4);
#pragma unroll
  for (int nt = 0; nt < 4; ++nt) {
    AccU u; u.v = acc[nt];
    const int t = t0 + nt * 16 + tl;
#pragma unroll
    for (int j = 0; j < 8; ++j) {
      int o = o0 + w * 16 + ob + j;
      float val = u.f[j] + qkv_b[o];
      if (o < 256) {                       // q (scaled)
        val *= 0.17677669529663687f;       // DKH^-0.5
        int h = o >> 5, d = o & 31;
        qT[(((size_t)b * NHh + h) * Tt + t) * DKH + d] = f2bf(val);
      } else if (o < 512) {                // k
        int oo = o - 256, h = oo >> 5, d = oo & 31;
        kT[(((size_t)b * NHh + h) * Tt + t) * DKH + d] = f2bf(val);
      } else {                             // v  (stored [d][t])
        int oo = o - 512, h = oo >> 5, d = oo & 31;
        vD[(((size_t)b * NHh + h) * DKH + d) * Tt + t] = f2bf(val);
      }
    }
  }
}

// ---------------------------------------------------------------------------
// K2: attention core per (b, h, 32-row t-block), one wave, two 16-row tiles.
// logits = qT*kT^T; rel WMMA tiles scatter-added via s = m - 255 + t
// (rel_to_abs is a per-row bijection, so no rel buffer is needed);
// softmax; attn = w*v.  Output transposed into attnT[b][t][c], c = h*32+d.
// ---------------------------------------------------------------------------
__global__ void k_attention(const unsigned short* __restrict__ qT,
                            const unsigned short* __restrict__ kT,
                            const unsigned short* __restrict__ vD,
                            const unsigned short* __restrict__ relw_bf,
                            unsigned short* __restrict__ attnT) {
  __shared__ float          s_log[32 * 256];   // 32 KB
  __shared__ unsigned short s_w[32 * 256];     // 16 KB

  const int lane = threadIdx.x;
  const int b = blockIdx.x, h = blockIdx.y, tz = blockIdx.z;
  const int t0 = tz * 32;                      // 32 query rows per workgroup
  const size_t bh = (size_t)b * NHh + h;
  const unsigned short* qTb = qT + bh * Tt * DKH;
  const unsigned short* kTb = kT + bh * Tt * DKH;
  const unsigned short* vDb = vD + bh * DKH * Tt;

  const int tl = lane & 15, rb = 8 * (lane >> 4);

  // q fragments for the two 16-row tiles (full K = DKH = 32)
  v16bf qf0 = load_a_frag(qTb + (size_t)t0 * DKH, DKH, lane);
  v16bf qf1 = load_a_frag(qTb + (size_t)(t0 + 16) * DKH, DKH, lane);

  // ---- logits: 16 s-tiles, k-fragment shared by both q tiles --------------
#pragma unroll
  for (int st = 0; st < 16; ++st) {
    v16bf bf = load_b_frag(kTb + (size_t)(st * 16) * DKH, DKH, lane);
    v8f z = {};
    v8f c0 = wmma_bf(qf0, bf, z);
    v8f c1 = wmma_bf(qf1, bf, z);
    AccU u0, u1; u0.v = c0; u1.v = c1;
    int col = st * 16 + tl;
#pragma unroll
    for (int j = 0; j < 8; ++j) {
      s_log[(rb + j) * 256 + col]      = u0.f[j];
      s_log[(16 + rb + j) * 256 + col] = u1.f[j];
    }
  }

  // ---- rel: 18 m-tiles cover m in [224-t0, 510-t0]; scatter-add -----------
  const int mt_lo = 14 - 2 * tz;               // (224 - t0) >> 4
#pragma unroll
  for (int mi = 0; mi < 18; ++mi) {
    int m0 = (mt_lo + mi) * 16;
    v16bf bf = load_b_frag(relw_bf + (size_t)m0 * DKH, DKH, lane);
    v8f z = {};
    v8f c0 = wmma_bf(qf0, bf, z);
    v8f c1 = wmma_bf(qf1, bf, z);
    AccU u0, u1; u0.v = c0; u1.v = c1;
    int mcol = m0 + tl;
#pragma unroll
    for (int j = 0; j < 8; ++j) {
      int r0 = rb + j;                          // t = t0 + r0
      int s0 = mcol - 255 + t0 + r0;            // rel_to_abs index
      if ((unsigned)s0 < 256u) s_log[r0 * 256 + s0] += u0.f[j];
      int r1 = 16 + rb + j;                     // t = t0 + r1
      int s1 = mcol - 255 + t0 + r1;
      if ((unsigned)s1 < 256u) s_log[r1 * 256 + s1] += u1.f[j];
    }
  }

  // ---- softmax (wave-cooperative, 8 cols per lane per row) ----------------
  for (int r = 0; r < 32; ++r) {
    float vals[8];
    float mx = -3.4e38f;
#pragma unroll
    for (int i = 0; i < 8; ++i) {
      float vv = s_log[r * 256 + lane + 32 * i];
      vals[i] = vv;
      mx = fmaxf(mx, vv);
    }
#pragma unroll
    for (int off = 16; off > 0; off >>= 1) mx = fmaxf(mx, __shfl_xor(mx, off, 32));
    float sum = 0.f;
#pragma unroll
    for (int i = 0; i < 8; ++i) { vals[i] = __expf(vals[i] - mx); sum += vals[i]; }
#pragma unroll
    for (int off = 16; off > 0; off >>= 1) sum += __shfl_xor(sum, off, 32);
    float inv = 1.f / sum;
#pragma unroll
    for (int i = 0; i < 8; ++i) s_w[r * 256 + lane + 32 * i] = f2bf(vals[i] * inv);
  }

  // ---- attn = w (2x16x256) * v^T (256x32): v-fragments shared -------------
#pragma unroll
  for (int dt = 0; dt < 2; ++dt) {
    v8f acc0 = {}, acc1 = {};
#pragma unroll
    for (int st = 0; st < 8; ++st) {
      v16bf bb = load_b_frag(vDb + (size_t)(dt * 16) * Tt + 32 * st, Tt, lane);
      v16bf a0 = load_a_frag((const unsigned short*)s_w + 32 * st, 256, lane);
      v16bf a1 = load_a_frag((const unsigned short*)s_w + 16 * 256 + 32 * st, 256, lane);
      acc0 = wmma_bf(a0, bb, acc0);
      acc1 = wmma_bf(a1, bb, acc1);
    }
    AccU u0, u1; u0.v = acc0; u1.v = acc1;
    int cidx = h * 32 + dt * 16 + tl;            // output channel
#pragma unroll
    for (int j = 0; j < 8; ++j) {
      attnT[((size_t)b * Tt + t0 + rb + j) * Cc + cidx]      = f2bf(u0.f[j]);
      attnT[((size_t)b * Tt + t0 + 16 + rb + j) * Cc + cidx] = f2bf(u1.f[j]);
    }
  }
}

// ---------------------------------------------------------------------------
// K3: out-projection + bias + residual + batchnorm affine + ReLU.
// grid (Bb, 4, 4), 128 threads, 64x64 block.
// ---------------------------------------------------------------------------
__global__ void k_out_gemm(const unsigned short* __restrict__ attnw_bf,
                           const unsigned short* __restrict__ attnT,
                           const float* __restrict__ attn_b,
                           const float* __restrict__ x,
                           const float* __restrict__ bn_gamma,
                           const float* __restrict__ bn_beta,
                           const float* __restrict__ bn_mean,
                           const float* __restrict__ bn_var,
                           float* __restrict__ out) {
  __shared__ unsigned short As[2 * 64 * 32];
  __shared__ unsigned short Bs[2 * 64 * 32];
  const int tid = threadIdx.x, lane = tid & 31, w = tid >> 5;
  const int b = blockIdx.x, o0 = blockIdx.y * 64, t0 = blockIdx.z * 64;

  v8f acc[4] = {};
  gemm64x64(attnw_bf + (size_t)o0 * Cc, attnT + ((size_t)b * Tt + t0) * Cc,
            As, Bs, tid, lane, w, acc);

  const int tl = lane & 15, ob = 8 * (lane >> 4);
  const int n = b / Vv, v = b - n * Vv;
#pragma unroll
  for (int nt = 0; nt < 4; ++nt) {
    AccU u; u.v = acc[nt];
    const int t = t0 + nt * 16 + tl;
#pragma unroll
    for (int j = 0; j < 8; ++j) {
      int o = o0 + w * 16 + ob + j;
      size_t xi = (((size_t)n * Cc + o) * Tt + t) * Vv + v;
      float inv = bn_gamma[o] * rsqrtf(bn_var[o] + 1e-5f);
      float r = (u.f[j] + attn_b[o] + x[xi]) * inv + (bn_beta[o] - bn_mean[o] * inv);
      out[xi] = fmaxf(r, 0.f);
    }
  }
}

// ---------------------------------------------------------------------------
extern "C" void kernel_launch(void* const* d_in, const int* in_sizes, int n_in,
                              void* d_out, int out_size, void* d_ws, size_t ws_size,
                              hipStream_t stream) {
  const float* x       = (const float*)d_in[0];
  const float* qkv_w   = (const float*)d_in[1];
  const float* qkv_b   = (const float*)d_in[2];
  const float* key_rel = (const float*)d_in[3];
  const float* attn_w  = (const float*)d_in[4];
  const float* attn_b  = (const float*)d_in[5];
  const float* bn_g    = (const float*)d_in[6];
  const float* bn_b    = (const float*)d_in[7];
  const float* bn_m    = (const float*)d_in[8];
  const float* bn_v    = (const float*)d_in[9];
  float* out = (float*)d_out;

  unsigned char* ws = (unsigned char*)d_ws;
  size_t off = 0;
  auto alloc = [&](size_t bytes) -> void* {
    void* p = ws + off;
    off += (bytes + 255) & ~(size_t)255;
    return p;
  };
  unsigned short* xT      = (unsigned short*)alloc((size_t)Bb * Tt * Cc * 2);   // 26.2 MB
  unsigned short* qkvw_bf = (unsigned short*)alloc((size_t)O3 * Cc * 2);
  unsigned short* attnw_bf= (unsigned short*)alloc((size_t)Cc * Cc * 2);
  unsigned short* relw_bf = (unsigned short*)alloc((size_t)512 * DKH * 2);
  unsigned short* qT      = (unsigned short*)alloc((size_t)Bb * NHh * Tt * DKH * 2);
  unsigned short* kT      = (unsigned short*)alloc((size_t)Bb * NHh * Tt * DKH * 2);
  unsigned short* vD      = (unsigned short*)alloc((size_t)Bb * NHh * DKH * Tt * 2);
  unsigned short* attnT   = xT;   // xT is dead after K1; reuse (stream-ordered)

  // K0: packing
  k_pack_x<<<(Bb * Tt * Cc) / 256, 256, 0, stream>>>(x, xT);
  int wtot = O3 * Cc + Cc * Cc + 512 * DKH;
  k_pack_w<<<(wtot + 255) / 256, 256, 0, stream>>>(qkv_w, attn_w, key_rel,
                                                   qkvw_bf, attnw_bf, relw_bf);
  // K1: QKV projection (200 x 12 x 4 blocks of 64x64, 4 waves each)
  k_qkv_gemm<<<dim3(Bb, O3 / 64, Tt / 64), 128, 0, stream>>>(
      qkvw_bf, xT, qkv_b, qT, kT, vD);
  // K2: attention core (200 x 8 x 8 blocks of 32 rows, 100 WMMA each)
  k_attention<<<dim3(Bb, NHh, Tt / 32), 32, 0, stream>>>(
      qT, kT, vD, relw_bf, attnT);
  // K3: output projection + residual + BN + ReLU (200 x 4 x 4 blocks)
  k_out_gemm<<<dim3(Bb, Cc / 64, Tt / 64), 128, 0, stream>>>(
      attnw_bf, attnT, attn_b, x, bn_g, bn_b, bn_m, bn_v, out);
}
